// Reservoir_47081431499437
// MI455X (gfx1250) — matrix operand order repaired and checked by reading, not compiled, not run on hardware
//
#include <hip/hip_runtime.h>
#include <math.h>

// ---------------------------------------------------------------------------
// ESN forward, teacher-forced:
//   s[n] = tanh(Wc @ [s[n-1]; x[n]; y[n-1]]) + 1e-3*(u[n]-0.5)
// Wc = [W | W_in | W_feedb | 0-pad] : 2048 x 2176  (68 chunks of K=32; the
// last chunk is zero padding so every wave runs a compile-time 17-chunk loop).
// Weights live in LDS (136 KB/WG over 64 WGs) -> per-step global traffic is
// only the 8 KB state broadcast; each step = 17 unrolled v_wmma per wave
// (two accumulator chains) + one device-scope barrier.
// ---------------------------------------------------------------------------

#define N_RES 2048
#define N_IN 64
#define N_OUT 32
#define K_TOT 2144              // 2048 + 64 + 32 (valid K)
#define K_PAD 2176              // padded to 68 chunks of 32
#define NCH 68                  // K_PAD / 32
#define CH_PER_WAVE 17          // 68 / 4 K-quarters, compile-time constant
#define NWG 64
#define ROWS_PER_WG 32          // N_RES / NWG
#define WG_THREADS 256
#define WSLAB_BYTES (2 * NCH * 1024)            // 2 row-tiles * 68 chunks * 1KB = 139264
#define SMEM_BYTES (WSLAB_BYTES + K_PAD * 2 + ROWS_PER_WG * 4)   // 143744
#define NOISE_AMP 0.001f

typedef __attribute__((ext_vector_type(16))) __bf16 v16bf;
typedef __attribute__((ext_vector_type(8)))  float  v8f;

static __device__ __forceinline__ unsigned short f2bf(float f) {
  unsigned u = __builtin_bit_cast(unsigned, f);
  unsigned r = u + 0x7FFFu + ((u >> 16) & 1u);   // round-to-nearest-even
  return (unsigned short)(r >> 16);
}

// ---------------------------------------------------------------------------
// Prep: fold [W | W_in | W_fb | 0] -> bf16, swizzled to the WMMA 16-bit
// A-matrix per-lane layout (ISA 7.12.2): lane m (0-15): V0..3 = K0..7,
// V4..7 = K16..23; lane m+16: V0..3 = K8..15, V4..7 = K24..31.  Stored so each
// lane's chunk is 32 contiguous bytes.  Also zeroes state buf0 + counter.
// ---------------------------------------------------------------------------
__global__ void esn_prep(const float* __restrict__ w, const float* __restrict__ w_in,
                         const float* __restrict__ w_fb, unsigned* __restrict__ swz,
                         unsigned* __restrict__ counter, float* __restrict__ sbuf0) {
  unsigned d = blockIdx.x * blockDim.x + threadIdx.x;
  if (d == 0) *counter = 0u;
  if (d < N_RES) sbuf0[d] = 0.0f;
  const unsigned DW = (N_RES * (unsigned)K_PAD) / 2u;   // packed dwords
  if (d >= DW) return;

  unsigned j  = d & 7u;          // VGPR slot 0..7
  unsigned li = d >> 3;
  unsigned l  = li & 31u;        // lane
  unsigned ci = li >> 5;
  unsigned c  = ci % NCH;        // K-chunk
  unsigned ti = ci / NCH;        // g*2 + t
  unsigned t  = ti & 1u;
  unsigned g  = ti >> 1;
  unsigned row = g * ROWS_PER_WG + t * 16u + (l & 15u);
  unsigned h  = l >> 4;
  unsigned kl = (j < 4u) ? (h * 8u + 2u * j) : (16u + h * 8u + 2u * (j - 4u));
  unsigned k0 = c * 32u + kl;

  float f0 = 0.0f, f1 = 0.0f;
  {
    unsigned k = k0;
    if (k < (unsigned)N_RES)                 f0 = w[(size_t)row * N_RES + k];
    else if (k < (unsigned)(N_RES + N_IN))   f0 = w_in[(size_t)row * N_IN + (k - N_RES)];
    else if (k < (unsigned)K_TOT)            f0 = w_fb[(size_t)row * N_OUT + (k - N_RES - N_IN)];
    k = k0 + 1u;
    if (k < (unsigned)N_RES)                 f1 = w[(size_t)row * N_RES + k];
    else if (k < (unsigned)(N_RES + N_IN))   f1 = w_in[(size_t)row * N_IN + (k - N_RES)];
    else if (k < (unsigned)K_TOT)            f1 = w_fb[(size_t)row * N_OUT + (k - N_RES - N_IN)];
  }
  swz[d] = (unsigned)f2bf(f0) | ((unsigned)f2bf(f1) << 16);
}

// ---------------------------------------------------------------------------
// Persistent recurrence kernel.  64 WGs x 256 threads (8 wave32).
// Wave w: row-tile t = w&1 (16 rows), K-quarter kq = w>>1 (exactly 17 chunks).
// B operand = state chunk replicated across all 16 N-columns -> two 16B
// LDS loads per lane (lanes 0-15: K0-7,16-23; lanes 16-31: K8-15,24-31).
// Fully unrolled chunk loop + dual accumulators so ds_loads pipeline ahead of
// the WMMA chain.  Device-scope spin barrier between steps.
// ---------------------------------------------------------------------------
__global__ void __launch_bounds__(WG_THREADS)
esn_run(const unsigned* __restrict__ swz,
        const float* __restrict__ inputs, const float* __restrict__ outputs,
        const float* __restrict__ noise_u,
        float* __restrict__ sbuf0, float* __restrict__ sbuf1,
        unsigned* __restrict__ counter, int nsteps) {
  extern __shared__ char smem[];
  unsigned short* wlds  = (unsigned short*)smem;                       // 139264 B
  unsigned short* svec  = (unsigned short*)(smem + WSLAB_BYTES);       // 4352 B
  float*          accum = (float*)(smem + WSLAB_BYTES + K_PAD * 2);    // 128 B

  const unsigned g    = blockIdx.x;
  const unsigned tid  = threadIdx.x;
  const unsigned lane = tid & 31u;
  const unsigned wave = tid >> 5;
  const unsigned t    = wave & 1u;
  const unsigned kq   = wave >> 1;
  const unsigned h    = lane >> 4;
  const unsigned c0   = kq * (unsigned)CH_PER_WAVE;

  // One-time: pin this WG's weight slab in LDS (linear 136 KB copy).
  {
    const uint4* src = (const uint4*)swz + (size_t)g * (WSLAB_BYTES / 16);
    uint4*       dst = (uint4*)wlds;
    for (unsigned i = tid; i < WSLAB_BYTES / 16; i += WG_THREADS) dst[i] = src[i];
  }
  __syncthreads();

  const unsigned short* wbase = wlds + (size_t)t * NCH * 512u + (size_t)lane * 16u
                                     + (size_t)c0 * 512u;
  const unsigned short* bbase = svec + (size_t)c0 * 32u + h * 8u;

  for (int i = 0; i < nsteps; ++i) {
    const int n = i + 1;
    const float* sin_b  = (i & 1) ? sbuf1 : sbuf0;
    float*       sout_b = (i & 1) ? sbuf0 : sbuf1;

    // Build combined bf16 vector [s[n-1]; x[n]; y[n-1]; 0-pad] in LDS.
    for (unsigned k = tid; k < (unsigned)K_PAD; k += WG_THREADS) {
      float v = 0.0f;
      if (k < (unsigned)N_RES)
        v = __hip_atomic_load(&sin_b[k], __ATOMIC_RELAXED, __HIP_MEMORY_SCOPE_AGENT);
      else if (k < (unsigned)(N_RES + N_IN))
        v = inputs[(size_t)n * N_IN + (k - N_RES)];
      else if (k < (unsigned)K_TOT)
        v = outputs[(size_t)i * N_OUT + (k - N_RES - N_IN)];
      svec[k] = f2bf(v);
    }
    if (tid < ROWS_PER_WG) accum[tid] = 0.0f;
    __syncthreads();

    // 17 x (16x16x32 bf16 WMMA), unrolled, two accumulator chains.
    v8f acc0 = {0.f, 0.f, 0.f, 0.f, 0.f, 0.f, 0.f, 0.f};
    v8f acc1 = {0.f, 0.f, 0.f, 0.f, 0.f, 0.f, 0.f, 0.f};
#pragma unroll
    for (int cc = 0; cc < CH_PER_WAVE; ++cc) {
      v16bf a = *(const v16bf*)(wbase + (size_t)cc * 512u);
      typedef __attribute__((ext_vector_type(8))) __bf16 v8bf_t;
      v8bf_t blo = *(const v8bf_t*)(bbase + (size_t)cc * 32u);
      v8bf_t bhi = *(const v8bf_t*)(bbase + (size_t)cc * 32u + 16u);
      v16bf b = __builtin_shufflevector(blo, bhi, 0, 1, 2, 3, 4, 5, 6, 7, 8, 9, 10, 11, 12, 13, 14, 15);
      if (cc & 1)
        acc1 = __builtin_amdgcn_wmma_f32_16x16x32_bf16(false, a, false, b,
                                                       (short)0, acc1, false, false);
      else
        acc0 = __builtin_amdgcn_wmma_f32_16x16x32_bf16(false, a, false, b,
                                                       (short)0, acc0, false, false);
    }
    v8f acc = acc0 + acc1;

    // Column 0 of D lives in lanes 0 (M=j) and 16 (M=8+j); reduce K-quarters.
    if (lane == 0u || lane == 16u) {
#pragma unroll
      for (int j = 0; j < 8; ++j)
        atomicAdd(&accum[t * 16u + h * 8u + (unsigned)j], acc[j]);
    }
    __syncthreads();

    if (tid < ROWS_PER_WG) {
      unsigned row = g * ROWS_PER_WG + tid;
      float ns = tanhf(accum[tid]) +
                 NOISE_AMP * (noise_u[(size_t)n * N_RES + row] - 0.5f);
      __hip_atomic_store(&sout_b[row], ns, __ATOMIC_RELAXED, __HIP_MEMORY_SCOPE_AGENT);
    }

    // Device-scope grid barrier (monotonic counter; 64*4096 << 2^32).
    __threadfence();
    __syncthreads();
    if (tid == 0) {
      __hip_atomic_fetch_add(counter, 1u, __ATOMIC_ACQ_REL, __HIP_MEMORY_SCOPE_AGENT);
      const unsigned target = (unsigned)(i + 1) * (unsigned)NWG;
      while (__hip_atomic_load(counter, __ATOMIC_ACQUIRE, __HIP_MEMORY_SCOPE_AGENT) < target)
        __builtin_amdgcn_s_sleep(2);
    }
    __syncthreads();
  }
}

// ---------------------------------------------------------------------------
// Readout: out[0:32] = out_w @ s_last + out_b ; out[32:2080] = s_last.
// ---------------------------------------------------------------------------
__global__ void esn_readout(const float* __restrict__ state, const float* __restrict__ out_w,
                            const float* __restrict__ out_b, float* __restrict__ out) {
  __shared__ float red[256];
  const unsigned r = blockIdx.x;      // 0..31
  const unsigned tid = threadIdx.x;
  float s = 0.f;
  for (unsigned k = tid; k < (unsigned)N_RES; k += 256u)
    s += out_w[(size_t)r * N_RES + k] * state[k];
  red[tid] = s;
  __syncthreads();
  for (unsigned off = 128; off > 0; off >>= 1) {
    if (tid < off) red[tid] += red[tid + off];
    __syncthreads();
  }
  if (tid == 0) out[r] = red[0] + out_b[r];
  if (tid < 64) out[N_OUT + r * 64u + tid] = state[r * 64u + tid];
}

// ---------------------------------------------------------------------------
extern "C" void kernel_launch(void* const* d_in, const int* in_sizes, int n_in,
                              void* d_out, int out_size, void* d_ws, size_t ws_size,
                              hipStream_t stream) {
  const float* inputs  = (const float*)d_in[0];
  const float* outputs = (const float*)d_in[1];
  const float* w       = (const float*)d_in[2];
  const float* w_in    = (const float*)d_in[3];
  const float* w_fb    = (const float*)d_in[4];
  const float* out_w   = (const float*)d_in[5];
  const float* out_b   = (const float*)d_in[6];
  const float* noise_u = (const float*)d_in[7];
  float* out = (float*)d_out;

  const int T = in_sizes[0] / N_IN;
  const int nsteps = T - 1;

  char* ws = (char*)d_ws;
  unsigned* counter = (unsigned*)ws;                         // 256 B slot
  float*    sbuf0   = (float*)(ws + 256);
  float*    sbuf1   = (float*)(ws + 256 + (size_t)N_RES * 4);
  unsigned* swz     = (unsigned*)(ws + 256 + (size_t)2 * N_RES * 4);  // ~8.9 MB

  (void)hipFuncSetAttribute(reinterpret_cast<const void*>(esn_run),
                            hipFuncAttributeMaxDynamicSharedMemorySize, SMEM_BYTES);

  const unsigned DW = (N_RES * (unsigned)K_PAD) / 2u;
  esn_prep<<<(DW + 255u) / 256u, 256, 0, stream>>>(w, w_in, w_fb, swz, counter, sbuf0);
  esn_run<<<NWG, WG_THREADS, SMEM_BYTES, stream>>>(swz, inputs, outputs, noise_u,
                                                   sbuf0, sbuf1, counter, nsteps);
  const float* last = (nsteps & 1) ? sbuf1 : sbuf0;
  esn_readout<<<N_OUT, 256, 0, stream>>>(last, out_w, out_b, out);
}
